// FeatureExtractor_61581241090434
// MI455X (gfx1250) — matrix operand order repaired and checked by reading, not compiled, not run on hardware
//
#include <hip/hip_runtime.h>
#include <stdint.h>

#define DEV __device__ __forceinline__

typedef __attribute__((ext_vector_type(16))) __bf16 v16bf;
typedef __attribute__((ext_vector_type(8)))  float  v8f;

union FragBF { v16bf v; float4 f4[2]; };

DEV v8f v8f_zero(){ v8f z; for (int i = 0; i < 8; ++i) z[i] = 0.0f; return z; }

DEV uint16_t f2bf(float x){
  uint32_t u = __float_as_uint(x);
  uint32_t r = (u + 0x7FFFu + ((u >> 16) & 1u)) >> 16;
  return (uint16_t)r;
}
DEV uint32_t pk2(float a, float b){ return (uint32_t)f2bf(a) | ((uint32_t)f2bf(b) << 16); }

// A fragment (16x32 bf16): lane 16*hi+m holds K = k0+8hi..+7 (VGPR0-3) and k0+16+8hi..+7 (VGPR4-7)
DEV v16bf load_frag_a(const uint16_t* base, int ldk, int k0, int lane){
  int m = lane & 15, hi = lane >> 4;
  const uint16_t* p = base + (size_t)m * ldk + k0 + hi * 8;
  FragBF f;
  f.f4[0] = *(const float4*)(p);
  f.f4[1] = *(const float4*)(p + 16);
  return f.v;
}
// B fragment (32x16 bf16) from a B^T (NxK, K contiguous) tile: lane 16*hi+n holds K = k0+8hi..+7 (VGPR0-3) and k0+16+8hi..+7 (VGPR4-7)
DEV v16bf load_frag_b(const uint16_t* bT, int ldk, int k0, int lane){
  int n = lane & 15, hi = lane >> 4;
  const uint16_t* p = bT + (size_t)n * ldk + k0 + hi * 8;
  FragBF f;
  f.f4[0] = *(const float4*)(p);
  f.f4[1] = *(const float4*)(p + 16);
  return f.v;
}
DEV v8f wmma_bf16(v16bf a, v16bf b, v8f c){
  return __builtin_amdgcn_wmma_f32_16x16x32_bf16(false, a, false, b, (short)0, c, false, false);
}

// CDNA5 async global->LDS copy (ASYNCcnt tracked, bypasses VGPRs)
DEV void async_b128(uint32_t lds_off, const void* gaddr){
  asm volatile("global_load_async_to_lds_b128 %0, %1, off"
               :: "v"(lds_off), "v"(gaddr) : "memory");
}
DEV void wait_async_le8(){ asm volatile("s_wait_asynccnt 0x8" ::: "memory"); }
DEV void wait_async_0(){  asm volatile("s_wait_asynccnt 0x0" ::: "memory"); }

// ---------------------------------------------------------------- weight packing
// conv weights (F,F',5,1) -> wp[f][kk*64+f'] row-major 64x320 bf16
__global__ void k_pack_convw(const float* __restrict__ w, uint16_t* __restrict__ wp){
  int i = blockIdx.x * 256 + threadIdx.x;
  if (i >= 64 * 320) return;
  int f = i / 320, r = i % 320;
  int kk = r / 64, fp = r % 64;
  wp[f * 320 + kk * 64 + fp] = f2bf(w[(f * 64 + fp) * 5 + kk]);
}
// wq/wk/wv/wv1 (64,192 with fi=f*3+slot) -> stacked 256x192, K reordered to slot*64+f
__global__ void k_pack_wqkv(const float* __restrict__ wq, const float* __restrict__ wk,
                            const float* __restrict__ wv, const float* __restrict__ wv1,
                            uint16_t* __restrict__ wp){
  int i = blockIdx.x * 256 + threadIdx.x;
  if (i >= 4 * 64 * 192) return;
  int p = i / (64 * 192); int r = i % (64 * 192);
  int d = r / 192; int q = r % 192; int slot = q / 64, f = q % 64;
  const float* src = (p == 0) ? wq : (p == 1) ? wk : (p == 2) ? wv : wv1;
  wp[(p * 64 + d) * 192 + slot * 64 + f] = f2bf(src[d * 192 + f * 3 + slot]);
}
__global__ void k_pack_whh(const float* __restrict__ whh, uint16_t* __restrict__ wp){
  int i = blockIdx.x * 256 + threadIdx.x;
  if (i >= 1536 * 512) return;
  wp[i] = f2bf(whh[i]);
}
// wih (1536, 8192 with i=d*128+k2) -> K reordered to k2*64+d (matches oG layout)
__global__ void k_pack_wih(const float* __restrict__ wih, uint16_t* __restrict__ wp){
  int i = blockIdx.x * 256 + threadIdx.x;
  if (i >= 1536 * 8192) return;
  int g = i >> 13; int r = i & 8191; int k2 = r >> 6; int d = r & 63;
  wp[i] = f2bf(wih[(size_t)g * 8192 + d * 128 + k2]);
}

// ---------------------------------------------------------------- conv1 (1->64 ch), output layout [b][l][c][f]
__global__ __launch_bounds__(256) void k_conv1(const float* __restrict__ x, const float* __restrict__ w1,
                                               const float* __restrict__ b1, uint16_t* __restrict__ h1){
  int idx = blockIdx.x * 256 + threadIdx.x;      // over 32*240*128
  if (idx >= 32 * 240 * 128) return;
  int c = idx & 127; int l = (idx >> 7) % 240; int b = idx / (240 * 128);
  float xv[5];
  for (int k = 0; k < 5; ++k) xv[k] = x[((size_t)(b * 244 + l + k)) * 128 + c];
  uint32_t outp[32];
  for (int f = 0; f < 64; f += 2){
    float a0 = b1[f], a1 = b1[f + 1];
    for (int k = 0; k < 5; ++k){ a0 += w1[f * 5 + k] * xv[k]; a1 += w1[(f + 1) * 5 + k] * xv[k]; }
    outp[f >> 1] = pk2(fmaxf(a0, 0.0f), fmaxf(a1, 0.0f));
  }
  uint4* dst = (uint4*)(h1 + (size_t)idx * 64);
  #pragma unroll
  for (int i = 0; i < 8; ++i) dst[i] = ((uint4*)outp)[i];
}

// ---------------------------------------------------------------- conv2/3/4 as WMMA GEMM (M=64, N=128, K=320)
// LDS: A 64x328 (320 used), Bt 128x328 (im2col^T, K contiguous)
__global__ __launch_bounds__(256) void k_conv_gemm(const uint16_t* __restrict__ hin, uint16_t* __restrict__ hout,
                                                   const uint16_t* __restrict__ wp, const float* __restrict__ bias,
                                                   int Lin, int Lout){
  extern __shared__ char smem[];
  uint16_t* AL  = (uint16_t*)smem;                       // 64 x 328
  uint16_t* BtL = (uint16_t*)(smem + 64 * 328 * 2);      // 128 x 328
  int blk = blockIdx.x; int l = blk % Lout; int b = blk / Lout;
  int tid = threadIdx.x;
  for (int i = tid; i < 64 * 40; i += 256){
    int r = i / 40, j = i % 40;
    ((float4*)(AL + r * 328))[j] = ((const float4*)(wp + r * 320))[j];
  }
  for (int q = tid; q < 640; q += 256){                   // 128 c x 5 kk chunks of 64 elements
    int kk = q >> 7; int c = q & 127;
    const float4* src = (const float4*)(hin + ((size_t)(b * Lin + l + kk) * 128 + c) * 64);
    float4* dst = (float4*)(BtL + c * 328 + kk * 64);
    #pragma unroll
    for (int i = 0; i < 8; ++i) dst[i] = src[i];
  }
  __syncthreads();
  int wave = tid >> 5, lane = tid & 31;
  int n0 = wave * 16;
  v8f acc[4]; for (int i = 0; i < 4; ++i) acc[i] = v8f_zero();
  for (int ks = 0; ks < 10; ++ks){
    int k0 = ks * 32;
    v16bf bf = load_frag_b(BtL + n0 * 328, 328, k0, lane);
    v16bf af[4];
    #pragma unroll
    for (int mt = 0; mt < 4; ++mt) af[mt] = load_frag_a(AL + mt * 16 * 328, 328, k0, lane);
    #pragma unroll
    for (int mt = 0; mt < 4; ++mt) acc[mt] = wmma_bf16(af[mt], bf, acc[mt]);
  }
  int n = lane & 15, hi = lane >> 4;
  int c = n0 + n;
  for (int mt = 0; mt < 4; ++mt){
    int f0 = mt * 16 + hi * 8;
    uint32_t pkd[4];
    #pragma unroll
    for (int e = 0; e < 8; e += 2){
      float v0 = fmaxf(acc[mt][e]     + bias[f0 + e],     0.0f);
      float v1 = fmaxf(acc[mt][e + 1] + bias[f0 + e + 1], 0.0f);
      pkd[e >> 1] = pk2(v0, v1);
    }
    *(uint4*)(hout + ((size_t)(b * Lout + l) * 128 + c) * 64 + f0) = *(uint4*)pkd;
  }
}

// ---------------------------------------------------------------- fused self-attention per (b,s)
// proj (M=256,K=192,N=128) -> S=f^T g (128x128,K=64) -> softmax over c -> o=v*beta (64x128,K=128) -> gamma*o+v1
__global__ __launch_bounds__(256) void k_attn(const uint16_t* __restrict__ h4, const uint16_t* __restrict__ wqkvP,
                                              const float* __restrict__ gamma, uint16_t* __restrict__ oG){
  extern __shared__ char smem[];
  uint16_t* BtL   = (uint16_t*)smem;                        // 128 x 200 (patch^T); later beta 128 x 136
  uint16_t* fT    = (uint16_t*)(smem + 51200);              // 128 x 72   [c][d]
  uint16_t* gT    = (uint16_t*)(smem + 51200 + 18432);      // 128 x 72   [c][d]
  uint16_t* vL    = (uint16_t*)(smem + 51200 + 36864);      // 64 x 136   [d][c]
  uint16_t* betaL = BtL;                                    // 128 x 136  [k2][c]
  int blk = blockIdx.x; int s = blk % 76; int b = blk / 76;
  int start = s * 3; if (start > 225) start = 225;
  int tid = threadIdx.x;
  for (int q = tid; q < 384; q += 256){                     // 128 c x 3 slot chunks of 64
    int slot = q >> 7; int c = q & 127;
    const float4* src = (const float4*)(h4 + ((size_t)(b * 228 + start + slot) * 128 + c) * 64);
    float4* dst = (float4*)(BtL + c * 200 + slot * 64);
    #pragma unroll
    for (int i = 0; i < 8; ++i) dst[i] = src[i];
  }
  __syncthreads();
  int wave = tid >> 5, lane = tid & 31;
  int n0 = wave * 16;
  int n = lane & 15, hi = lane >> 4;

  v8f acc[16]; for (int i = 0; i < 16; ++i) acc[i] = v8f_zero();
  for (int ks = 0; ks < 6; ++ks){
    int k0 = ks * 32;
    v16bf bf = load_frag_b(BtL + n0 * 200, 200, k0, lane);
    #pragma unroll
    for (int g4 = 0; g4 < 4; ++g4){
      v16bf af[4];
      #pragma unroll
      for (int t = 0; t < 4; ++t)
        af[t] = load_frag_a(wqkvP + (size_t)(g4 * 4 + t) * 16 * 192, 192, k0, lane);
      #pragma unroll
      for (int t = 0; t < 4; ++t)
        acc[g4 * 4 + t] = wmma_bf16(af[t], bf, acc[g4 * 4 + t]);
    }
  }
  int c = n0 + n;
  for (int mt = 0; mt < 4; ++mt){                           // f -> fT[c][d]
    uint32_t pkd[4];
    for (int e = 0; e < 8; e += 2) pkd[e >> 1] = pk2(acc[mt][e], acc[mt][e + 1]);
    *(uint4*)(fT + c * 72 + mt * 16 + hi * 8) = *(uint4*)pkd;
  }
  for (int mt = 4; mt < 8; ++mt){                           // g -> gT[c][d]
    uint32_t pkd[4];
    for (int e = 0; e < 8; e += 2) pkd[e >> 1] = pk2(acc[mt][e], acc[mt][e + 1]);
    *(uint4*)(gT + c * 72 + (mt - 4) * 16 + hi * 8) = *(uint4*)pkd;
  }
  for (int mt = 8; mt < 12; ++mt){                          // v -> vL[d][c]
    int d0 = (mt - 8) * 16 + hi * 8;
    for (int e = 0; e < 8; ++e) vL[(d0 + e) * 136 + c] = f2bf(acc[mt][e]);
  }
  v8f v1a[4]; for (int i = 0; i < 4; ++i) v1a[i] = acc[12 + i];
  __syncthreads();

  // S[c1][k2], wave owns k2-strip n0; K = 64 over d
  v8f accS[8]; for (int i = 0; i < 8; ++i) accS[i] = v8f_zero();
  for (int ks = 0; ks < 2; ++ks){
    int k0 = ks * 32;
    v16bf bf = load_frag_b(gT + n0 * 72, 72, k0, lane);
    #pragma unroll
    for (int g4 = 0; g4 < 2; ++g4){
      v16bf af[4];
      #pragma unroll
      for (int t = 0; t < 4; ++t)
        af[t] = load_frag_a(fT + (g4 * 4 + t) * 16 * 72, 72, k0, lane);
      #pragma unroll
      for (int t = 0; t < 4; ++t)
        accS[g4 * 4 + t] = wmma_bf16(af[t], bf, accS[g4 * 4 + t]);
    }
  }
  // softmax over c1: rows live across mt (in-lane) and the lane^16 half
  float mx = -1e30f;
  for (int mt = 0; mt < 8; ++mt) for (int e = 0; e < 8; ++e) mx = fmaxf(mx, accS[mt][e]);
  mx = fmaxf(mx, __shfl_xor(mx, 16, 32));
  float sum = 0.0f;
  for (int mt = 0; mt < 8; ++mt) for (int e = 0; e < 8; ++e){
    float t = __expf(accS[mt][e] - mx); accS[mt][e] = t; sum += t;
  }
  sum += __shfl_xor(sum, 16, 32);
  float inv = 1.0f / sum;
  __syncthreads();
  int k2 = n0 + n;
  for (int mt = 0; mt < 8; ++mt){                           // beta -> betaL[k2][c1]
    uint32_t pkd[4];
    for (int e = 0; e < 8; e += 2) pkd[e >> 1] = pk2(accS[mt][e] * inv, accS[mt][e + 1] * inv);
    *(uint4*)(betaL + k2 * 136 + mt * 16 + hi * 8) = *(uint4*)pkd;
  }
  __syncthreads();

  // o[d][k2] = sum_c v[d,c] beta[c,k2]; K = 128 over c
  v8f accO[4]; for (int i = 0; i < 4; ++i) accO[i] = v8f_zero();
  for (int ks = 0; ks < 4; ++ks){
    int k0 = ks * 32;
    v16bf bf = load_frag_b(betaL + n0 * 136, 136, k0, lane);
    v16bf af[4];
    #pragma unroll
    for (int mt = 0; mt < 4; ++mt) af[mt] = load_frag_a(vL + mt * 16 * 136, 136, k0, lane);
    #pragma unroll
    for (int mt = 0; mt < 4; ++mt) accO[mt] = wmma_bf16(af[mt], bf, accO[mt]);
  }
  float gm = gamma[0];
  for (int mt = 0; mt < 4; ++mt){
    int d0 = mt * 16 + hi * 8;
    uint32_t pkd[4];
    for (int e = 0; e < 8; e += 2){
      float o0 = gm * accO[mt][e]     + v1a[mt][e];
      float o1 = gm * accO[mt][e + 1] + v1a[mt][e + 1];
      pkd[e >> 1] = pk2(o0, o1);
    }
    // oG layout [s][b][k2][d] (K = k2*64+d contiguous for the xg GEMM)
    *(uint4*)(oG + ((size_t)(s * 32 + b) * 128 + k2) * 64 + d0) = *(uint4*)pkd;
  }
}

// ---------------------------------------------------------------- xg = seq @ wih^T + bih (M=1536, N=2432, K=8192)
// double-buffered LDS, staged with CDNA5 async global->LDS copies (ASYNCcnt)
__global__ __launch_bounds__(256) void k_gemm_xg(const uint16_t* __restrict__ wihP, const uint16_t* __restrict__ oG,
                                                 const float* __restrict__ bih, float* __restrict__ xg){
  extern __shared__ char smem[];
  // 4 tiles of 128 x 72 bf16 (A0,B0,A1,B1), 18432 B each
  uint16_t* tiles = (uint16_t*)smem;
  uint32_t  base  = (uint32_t)(uintptr_t)smem;  // LDS byte offset (low 32 bits of LDS aperture addr)
  int m0 = blockIdx.x * 128;
  int nb = blockIdx.y * 128;
  int tid = threadIdx.x;
  int wave = tid >> 5, lane = tid & 31;
  int n0 = wave * 16;

  // issue one staging pass (8 async b128 per thread: 4 for A, 4 for B)
  auto stage = [&](int bufsel, int k0){
    uint32_t aoff = base + (uint32_t)(bufsel * 36864);
    uint32_t boff = aoff + 18432u;
    for (int i = tid; i < 1024; i += 256){
      int r = i >> 3, j = i & 7;
      uint32_t dl = (uint32_t)(r * 144 + j * 16);
      async_b128(aoff + dl, wihP + (size_t)(m0 + r) * 8192 + k0 + j * 8);
      async_b128(boff + dl, oG   + (size_t)(nb + r) * 8192 + k0 + j * 8);
    }
  };

  v8f acc[8]; for (int i = 0; i < 8; ++i) acc[i] = v8f_zero();
  stage(0, 0);
  for (int kt = 0; kt < 128; ++kt){
    int cur = kt & 1;
    __syncthreads();                       // prior compute on the other buffer is done
    if (kt + 1 < 128){
      stage(cur ^ 1, (kt + 1) * 64);
      wait_async_le8();                    // async retires in order: stage kt complete, kt+1 in flight
    } else {
      wait_async_0();
    }
    __syncthreads();                       // every wave's staged data visible
    const uint16_t* AL = tiles + cur * 18432;   // element offsets (2B each)
    const uint16_t* BL = AL + 9216;
    #pragma unroll
    for (int ki = 0; ki < 2; ++ki){
      int kk = ki * 32;
      v16bf bf = load_frag_b(BL + n0 * 72, 72, kk, lane);
      v16bf af[8];
      #pragma unroll
      for (int mt = 0; mt < 8; ++mt) af[mt] = load_frag_a(AL + mt * 16 * 72, 72, kk, lane);
      #pragma unroll
      for (int mt = 0; mt < 8; ++mt) acc[mt] = wmma_bf16(af[mt], bf, acc[mt]);
    }
  }
  int n = lane & 15, hi = lane >> 4;
  size_t nn = (size_t)nb + n0 + n;
  for (int mt = 0; mt < 8; ++mt){
    int g0 = m0 + mt * 16 + hi * 8;
    float o[8];
    for (int e = 0; e < 8; ++e) o[e] = acc[mt][e] + bih[g0 + e];
    *(float4*)(xg + nn * 1536 + g0)     = *(float4*)(o);
    *(float4*)(xg + nn * 1536 + g0 + 4) = *(float4*)(o + 4);
  }
}

// ---------------------------------------------------------------- GRU recurrence
__global__ void k_zero_h(float* __restrict__ hF, uint16_t* __restrict__ hB){
  int i = blockIdx.x * 256 + threadIdx.x;
  if (i < 32 * 512){ hF[i] = 0.0f; hB[i] = 0; }
}
// hg = hprev @ whh^T + bhh  (M=1536, N=32, K=512), A/B fragments straight from global (L2-hot)
__global__ __launch_bounds__(256) void k_gru_hg(const uint16_t* __restrict__ whhP, const uint16_t* __restrict__ hB,
                                                const float* __restrict__ bhh, float* __restrict__ hg){
  int tid = threadIdx.x; int wave = tid >> 5, lane = tid & 31;
  int m0 = blockIdx.x * 128 + wave * 16;
  v8f a0 = v8f_zero(), a1 = v8f_zero();
  for (int ks = 0; ks < 16; ++ks){
    int k0 = ks * 32;
    v16bf af = load_frag_a(whhP + (size_t)m0 * 512, 512, k0, lane);
    v16bf b0 = load_frag_b(hB,            512, k0, lane);
    v16bf b1 = load_frag_b(hB + 16 * 512, 512, k0, lane);
    a0 = wmma_bf16(af, b0, a0);
    a1 = wmma_bf16(af, b1, a1);
  }
  int n = lane & 15, hi = lane >> 4;
  int g0 = m0 + hi * 8;
  float o[8];
  for (int e = 0; e < 8; ++e) o[e] = a0[e] + bhh[g0 + e];
  *(float4*)(hg + (size_t)n * 1536 + g0)     = *(float4*)(o);
  *(float4*)(hg + (size_t)n * 1536 + g0 + 4) = *(float4*)(o + 4);
  for (int e = 0; e < 8; ++e) o[e] = a1[e] + bhh[g0 + e];
  *(float4*)(hg + (size_t)(n + 16) * 1536 + g0)     = *(float4*)(o);
  *(float4*)(hg + (size_t)(n + 16) * 1536 + g0 + 4) = *(float4*)(o + 4);
}
__global__ void k_gru_update(const float* __restrict__ xg_s, const float* __restrict__ hg,
                             float* __restrict__ hF, uint16_t* __restrict__ hB, float* __restrict__ outs_s){
  int idx = blockIdx.x * 256 + threadIdx.x;     // 32*512
  int b = idx >> 9, h = idx & 511;
  float xr = xg_s[b * 1536 + h], xz = xg_s[b * 1536 + 512 + h], xn = xg_s[b * 1536 + 1024 + h];
  float hr = hg[b * 1536 + h],   hz = hg[b * 1536 + 512 + h],   hn = hg[b * 1536 + 1024 + h];
  float r = 1.0f / (1.0f + __expf(-(xr + hr)));
  float z = 1.0f / (1.0f + __expf(-(xz + hz)));
  float nn = tanhf(xn + r * hn);
  float hp = hF[idx];
  float hnew = (1.0f - z) * nn + z * hp;
  hF[idx] = hnew;
  hB[idx] = f2bf(hnew);
  outs_s[idx] = hnew;
}

// ---------------------------------------------------------------- attention pooling over s, per b
__global__ void k_pool(const float* __restrict__ outs, const float* __restrict__ fcw,
                       const float* __restrict__ fcb, float* __restrict__ dout){
  __shared__ float att[76];
  __shared__ float red[8];
  int b = blockIdx.x, tid = threadIdx.x;
  int wave = tid >> 5, lane = tid & 31;
  for (int s = 0; s < 76; ++s){
    float p = 0.0f;
    for (int h = tid; h < 512; h += 256) p += outs[((size_t)s * 32 + b) * 512 + h] * fcw[h];
    for (int off = 16; off; off >>= 1) p += __shfl_xor(p, off, 32);
    if (lane == 0) red[wave] = p;
    __syncthreads();
    if (tid == 0){ float t = 0.0f; for (int w = 0; w < 8; ++w) t += red[w]; att[s] = t + fcb[0]; }
    __syncthreads();
  }
  float mx = -1e30f;
  for (int s = 0; s < 76; ++s) mx = fmaxf(mx, att[s]);
  float sum = 0.0f;
  for (int s = 0; s < 76; ++s) sum += __expf(att[s] - mx);
  float inv = 1.0f / sum;
  for (int h = tid; h < 512; h += 256){
    float a = 0.0f;
    for (int s = 0; s < 76; ++s) a += __expf(att[s] - mx) * inv * outs[((size_t)s * 32 + b) * 512 + h];
    dout[b * 512 + h] = a;
  }
}

// ================================================================ host side
extern "C" void kernel_launch(void* const* d_in, const int* in_sizes, int n_in,
                              void* d_out, int out_size, void* d_ws, size_t ws_size,
                              hipStream_t stream) {
  (void)in_sizes; (void)n_in; (void)out_size; (void)ws_size;
  const float* x   = (const float*)d_in[0];
  const float* w1  = (const float*)d_in[1];  const float* b1 = (const float*)d_in[2];
  const float* w2  = (const float*)d_in[3];  const float* b2 = (const float*)d_in[4];
  const float* w3  = (const float*)d_in[5];  const float* b3 = (const float*)d_in[6];
  const float* w4  = (const float*)d_in[7];  const float* b4 = (const float*)d_in[8];
  const float* wq  = (const float*)d_in[9];  const float* wk = (const float*)d_in[10];
  const float* wv  = (const float*)d_in[11]; const float* wv1 = (const float*)d_in[12];
  const float* gm  = (const float*)d_in[13];
  const float* wih = (const float*)d_in[14]; const float* whh = (const float*)d_in[15];
  const float* bih = (const float*)d_in[16]; const float* bhh = (const float*)d_in[17];
  const float* fcw = (const float*)d_in[18]; const float* fcb = (const float*)d_in[19];

  char* ws = (char*)d_ws;
  size_t off = 0;
  auto take = [&](size_t bytes)->char* {
    char* p = ws + off;
    off += (bytes + 255) & ~(size_t)255;
    return p;
  };
  char* bufA = take(125829120ull);          // h1 / h3 / oG
  char* bufB = take(123731968ull);          // h2 / h4 / xg
  uint16_t* w2p   = (uint16_t*)take(40960);
  uint16_t* w3p   = (uint16_t*)take(40960);
  uint16_t* w4p   = (uint16_t*)take(40960);
  uint16_t* wqkvP = (uint16_t*)take(98304);
  uint16_t* wihP  = (uint16_t*)take(25165824ull);
  uint16_t* whhP  = (uint16_t*)take(1572864ull);
  float*    hg    = (float*)take(196608);
  float*    hF    = (float*)take(65536);
  uint16_t* hB    = (uint16_t*)take(32768);
  float*    outs  = (float*)take(4980736ull);

  uint16_t* h1 = (uint16_t*)bufA;
  uint16_t* h2 = (uint16_t*)bufB;
  uint16_t* h3 = (uint16_t*)bufA;
  uint16_t* h4 = (uint16_t*)bufB;
  uint16_t* oG = (uint16_t*)bufA;
  float*    xg = (float*)bufB;

  // weight packing
  k_pack_convw<<<80, 256, 0, stream>>>(w2, w2p);
  k_pack_convw<<<80, 256, 0, stream>>>(w3, w3p);
  k_pack_convw<<<80, 256, 0, stream>>>(w4, w4p);
  k_pack_wqkv<<<192, 256, 0, stream>>>(wq, wk, wv, wv1, wqkvP);
  k_pack_whh<<<3072, 256, 0, stream>>>(whh, whhP);
  k_pack_wih<<<49152, 256, 0, stream>>>(wih, wihP);

  // conv stack
  k_conv1<<<3840, 256, 0, stream>>>(x, w1, b1, h1);
  k_conv_gemm<<<32 * 236, 256, 125952, stream>>>(h1, h2, w2p, b2, 240, 236);
  k_conv_gemm<<<32 * 232, 256, 125952, stream>>>(h2, h3, w3p, b3, 236, 232);
  k_conv_gemm<<<32 * 228, 256, 125952, stream>>>(h3, h4, w4p, b4, 232, 228);

  // fused attention per (b,s)
  k_attn<<<32 * 76, 256, 105472, stream>>>(h4, wqkvP, gm, oG);

  // GRU input projection (async double-buffered)
  k_gemm_xg<<<dim3(12, 19, 1), 256, 73728, stream>>>(wihP, oG, bih, xg);

  // sequential GRU
  k_zero_h<<<64, 256, 0, stream>>>(hF, hB);
  for (int s = 0; s < 76; ++s){
    k_gru_hg<<<12, 256, 0, stream>>>(whhP, hB, bhh, hg);
    k_gru_update<<<64, 256, 0, stream>>>(xg + (size_t)s * 32 * 1536, hg, hF, hB,
                                         outs + (size_t)s * 32 * 512);
  }

  // attention pooling -> (32, 512)
  k_pool<<<32, 256, 0, stream>>>(outs, fcw, fcb, (float*)d_out);
}
